// MoNetUnet_87299505258948
// MI455X (gfx1250) — matrix operand order repaired and checked by reading, not compiled, not run on hardware
//
#include <hip/hip_runtime.h>

// ---------------- mesh constants (from reference) ----------------
#define NV7 163842
#define NV6 40962
#define NV5 10242
#define NE7 983040
#define NE6 245760
#define NE5 61440

typedef __attribute__((ext_vector_type(16))) __bf16 v16bf;
typedef __attribute__((ext_vector_type(8)))  __bf16 v8bf;
typedef __attribute__((ext_vector_type(8)))  float  v8f;

__device__ __forceinline__ __bf16 f2bf(float f) {
  union { float f; unsigned u; } v; v.f = f;
  unsigned r = v.u + 0x7fffu + ((v.u >> 16) & 1u);   // round-to-nearest-even
  unsigned short h = (unsigned short)(r >> 16);
  return __builtin_bit_cast(__bf16, h);
}

// f32 -> bf16 elementwise (activations), row-major preserved
__global__ __launch_bounds__(256) void convert_bf16_kernel(
    const float* __restrict__ x, __bf16* __restrict__ y, long long n) {
  long long i = (long long)blockIdx.x * blockDim.x + threadIdx.x;
  if (i < n) y[i] = f2bf(x[i]);
}

// Pack B[Kd,N] (f32 row-major) into per-lane WMMA fragment order (bf16):
// chunk index ((tN*KSTEPS + kstep)*32 + lane), 16 bf16 per chunk.
__global__ __launch_bounds__(256) void pack_b_kernel(
    const float* __restrict__ B, __bf16* __restrict__ frag, int Kd, int N) {
  int ksteps = Kd >> 5;
  int ntiles = (N + 15) >> 4;
  int total = ntiles * ksteps * 32;
  int t = blockIdx.x * blockDim.x + threadIdx.x;
  if (t >= total) return;
  int lane = t & 31;
  int rest = t >> 5;
  int kstep = rest % ksteps;
  int tN = rest / ksteps;
  int col = tN * 16 + (lane & 15);
  int khalf = lane >> 4;
  __bf16* out = frag + (size_t)t * 16;
#pragma unroll
  for (int j = 0; j < 8; ++j) {
    int k = kstep * 32 + 8 * khalf + 2 * j + (j >= 4 ? 8 : 0);
    float b0 = (col < N) ? B[(size_t)k * N + col] : 0.0f;
    float b1 = (col < N) ? B[(size_t)(k + 1) * N + col] : 0.0f;
    out[2 * j]     = f2bf(b0);
    out[2 * j + 1] = f2bf(b1);
  }
}

// C[M,NN] = A[M,Kd]*B[Kd,NN]; A bf16 row-major, B pre-packed fragments.
// Two 16x16 N-tiles per wave (A-fragment reuse); K loop fully unrolled.
template <int KSTEPS, int NN>
__global__ __launch_bounds__(128) void gemm_wmma_packed(
    const __bf16* __restrict__ A, const __bf16* __restrict__ Bfrag,
    float* __restrict__ C, int M) {
  constexpr int Kd = KSTEPS * 32;
  const int lane = threadIdx.x & 31;
  const int wave = threadIdx.x >> 5;
  const int tM = blockIdx.x;
  const int tN0 = (blockIdx.y * 4 + wave) * 2;
  if (tN0 * 16 >= NN) return;                  // wave-uniform; EXEC stays all-1s for WMMA
  const int m0 = tM * 16;
  int row = m0 + (lane & 15); if (row >= M) row = M - 1;  // clamp; stores guarded
  const int khalf = lane >> 4;
  const __bf16* Arow = A + (size_t)row * Kd + 8 * khalf;
  const __bf16* Bf0 = Bfrag + ((size_t)tN0 * KSTEPS * 32 + lane) * 16;
  const __bf16* Bf1 = Bf0 + (size_t)KSTEPS * 32 * 16;
  v8f acc0 = {}, acc1 = {};
#pragma unroll
  for (int s = 0; s < KSTEPS; ++s) {
    v8bf alo = *(const v8bf*)(Arow + s * 32);        // K = kb + 8*khalf + 0..7
    v8bf ahi = *(const v8bf*)(Arow + s * 32 + 16);   // K = kb + 16 + 8*khalf + 0..7
    v16bf a = __builtin_shufflevector(alo, ahi, 0, 1, 2, 3, 4, 5, 6, 7,
                                      8, 9, 10, 11, 12, 13, 14, 15);
    v16bf b0 = *(const v16bf*)(Bf0 + (size_t)s * 512);
    v16bf b1 = *(const v16bf*)(Bf1 + (size_t)s * 512);
    acc0 = __builtin_amdgcn_wmma_f32_16x16x32_bf16(false, a, false, b0, (short)0, acc0,
                                                   false, false);
    acc1 = __builtin_amdgcn_wmma_f32_16x16x32_bf16(false, a, false, b1, (short)0, acc1,
                                                   false, false);
  }
  // D: VGPR j -> m = m0 + j + 8*khalf; lanes 0-15 span n (coalesced), compile-time strides
  float* Cp0 = C + (size_t)(m0 + 8 * khalf) * NN + tN0 * 16 + (lane & 15);
  float* Cp1 = Cp0 + 16;
  if (m0 + 16 <= M) {                          // full tile: unguarded immediate-offset stores
#pragma unroll
    for (int j = 0; j < 8; ++j) {
      Cp0[(size_t)j * NN] = acc0[j];
      Cp1[(size_t)j * NN] = acc1[j];
    }
  } else {                                     // single M-edge tile: guarded
#pragma unroll
    for (int j = 0; j < 8; ++j) {
      if (m0 + 8 * khalf + j < M) {
        Cp0[(size_t)j * NN] = acc0[j];
        Cp1[(size_t)j * NN] = acc1[j];
      }
    }
  }
}

// per-edge gaussian mixture weights: wbuf[e,k] = exp(-0.5*sum_d diff^2/(eps+sg^2))
__global__ __launch_bounds__(256) void edge_w_kernel(
    const float* __restrict__ pseudo, const float* __restrict__ mu,
    const float* __restrict__ sg, float* __restrict__ wbuf, int E) {
  int e = blockIdx.x * blockDim.x + threadIdx.x;
  if (e >= E) return;
  float p0 = pseudo[2 * e], p1 = pseudo[2 * e + 1];
#pragma unroll
  for (int k = 0; k < 3; ++k) {
    float d0 = p0 - mu[2 * k],  d1 = p1 - mu[2 * k + 1];
    float s0 = sg[2 * k],       s1 = sg[2 * k + 1];
    wbuf[3 * e + k] =
        __expf(-0.5f * (d0 * d0 / (1e-15f + s0 * s0) + d1 * d1 / (1e-15f + s1 * s1)));
  }
}

// per-(edge,channel): m = sum_k xg[src,k,o]*w[k]; atomic add into agg[dst,o]
template <int O>
__global__ __launch_bounds__(256) void edge_msg_t(
    const float* __restrict__ xg, const int* __restrict__ src, const int* __restrict__ dst,
    const float* __restrict__ wbuf, float* __restrict__ agg, int E) {
  long long idx = (long long)blockIdx.x * blockDim.x + threadIdx.x;
  if (idx >= (long long)E * O) return;
  int e = (int)(idx / O);                     // O power of two -> shifts
  int o = (int)(idx & (O - 1));
  int s = src[e], d = dst[e];
  float w0 = wbuf[3 * e], w1 = wbuf[3 * e + 1], w2 = wbuf[3 * e + 2];
  const float* xs = xg + (size_t)s * 3 * O + o;
  float m = xs[0] * w0 + xs[O] * w1 + xs[2 * O] * w2;
  atomicAdd(&agg[(size_t)d * O + o], m);
}

__global__ __launch_bounds__(256) void deg_kernel(const int* __restrict__ dst,
                                                  float* __restrict__ deg, int E) {
  int e = blockIdx.x * blockDim.x + threadIdx.x;
  if (e < E) atomicAdd(&deg[dst[e]], 1.0f);
}

// out = relu(agg/max(deg,1) + rootp + bias)
__global__ __launch_bounds__(256) void finalize_kernel(
    const float* __restrict__ agg, const float* __restrict__ deg,
    const float* __restrict__ rootp, const float* __restrict__ bias,
    float* __restrict__ out, int N, int O) {
  long long idx = (long long)blockIdx.x * blockDim.x + threadIdx.x;
  if (idx >= (long long)N * O) return;
  int n = (int)(idx / O), o = (int)(idx % O);
  float v = agg[idx] / fmaxf(deg[n], 1.0f) + rootp[idx] + bias[o];
  out[idx] = fmaxf(v, 0.0f);
}

// HexPool: xout[n,c] = max_{j<7} xin[idx[n,j], c]
__global__ __launch_bounds__(256) void hexpool_kernel(
    const float* __restrict__ xin, const int* __restrict__ idx,
    float* __restrict__ xout, int Nc, int C) {
  long long t = (long long)blockIdx.x * blockDim.x + threadIdx.x;
  if (t >= (long long)Nc * C) return;
  int n = (int)(t / C), c = (int)(t % C);
  float m = -3.4e38f;
#pragma unroll
  for (int j = 0; j < 7; ++j) m = fmaxf(m, xin[(size_t)idx[n * 7 + j] * C + c]);
  xout[t] = m;
}

// HexUnpool + skip concat
__global__ __launch_bounds__(256) void unpool_concat_kernel(
    const float* __restrict__ xc, const int* __restrict__ up, const float* __restrict__ skip,
    float* __restrict__ out, int Nc, int Nf, int C1, int C2) {
  int CT = C1 + C2;
  long long t = (long long)blockIdx.x * blockDim.x + threadIdx.x;
  if (t >= (long long)Nf * CT) return;
  int n = (int)(t / CT), c = (int)(t % CT);
  float v;
  if (c < C1) {
    if (n < Nc) v = xc[(size_t)n * C1 + c];
    else {
      int a = up[(n - Nc) * 2], b = up[(n - Nc) * 2 + 1];
      v = 0.5f * (xc[(size_t)a * C1 + c] + xc[(size_t)b * C1 + c]);
    }
  } else {
    v = skip[(size_t)n * C2 + (c - C1)];
  }
  out[t] = v;
}

// logits = x@W + b (32->2), then log_softmax
__global__ __launch_bounds__(256) void fc_lsm_kernel(
    const float* __restrict__ x, const float* __restrict__ w, const float* __restrict__ b,
    float* __restrict__ out, int N) {
  int n = blockIdx.x * blockDim.x + threadIdx.x;
  if (n >= N) return;
  float l0 = b[0], l1 = b[1];
#pragma unroll
  for (int c = 0; c < 32; ++c) {
    float v = x[(size_t)n * 32 + c];
    l0 += v * w[c * 2];
    l1 += v * w[c * 2 + 1];
  }
  float m = fmaxf(l0, l1);
  float lse = m + __logf(__expf(l0 - m) + __expf(l1 - m));
  out[(size_t)n * 2]     = l0 - lse;
  out[(size_t)n * 2 + 1] = l1 - lse;
}

// ---------------- host orchestration ----------------
struct GmmWs {
  float *xg, *rootp, *agg, *deg, *wbuf;
  __bf16 *xbf, *gfrag, *rfrag;
};

static void launch_gemm(const __bf16* A, const __bf16* Bf, float* C,
                        int M, int Kd, int N, hipStream_t s) {
  dim3 blk(128);
  dim3 g((M + 15) / 16, (N + 127) / 128);     // 4 waves x 2 tiles = 128 cols per block
#define GEMM_CASE(KS, NN_)                                                     \
  if (Kd == (KS)*32 && N == (NN_)) {                                           \
    gemm_wmma_packed<KS, NN_><<<g, blk, 0, s>>>(A, Bf, C, M);                  \
    return;                                                                    \
  }
  GEMM_CASE(1, 96)  GEMM_CASE(1, 32)          // cin=32 layers
  GEMM_CASE(2, 192) GEMM_CASE(2, 64)          // cin=64
  GEMM_CASE(4, 384) GEMM_CASE(4, 128)         // cin=128
  GEMM_CASE(6, 192) GEMM_CASE(6, 64)          // cin=192
  GEMM_CASE(3, 96)  GEMM_CASE(3, 32)          // cin=96
#undef GEMM_CASE
}

static void run_gmm_layer(const float* x, int N, int cin, int cout,
                          const float* G, const float* mu, const float* sg,
                          const float* root, const float* bias,
                          const int* edges, const float* ev, int E,
                          float* out, const GmmWs& w, hipStream_t stream) {
  const int KO = 3 * cout;
  // activations -> bf16
  long long nx = (long long)N * cin;
  convert_bf16_kernel<<<(unsigned)((nx + 255) / 256), 256, 0, stream>>>(x, w.xbf, nx);
  // weights -> packed bf16 fragments
  int gtot = ((KO + 15) / 16) * (cin / 32) * 32;
  pack_b_kernel<<<(gtot + 255) / 256, 256, 0, stream>>>(G, w.gfrag, cin, KO);
  int rtot = ((cout + 15) / 16) * (cin / 32) * 32;
  pack_b_kernel<<<(rtot + 255) / 256, 256, 0, stream>>>(root, w.rfrag, cin, cout);
  // xg = x@G ; rootp = x@root (WMMA)
  launch_gemm(w.xbf, w.gfrag, w.xg, N, cin, KO, stream);
  launch_gemm(w.xbf, w.rfrag, w.rootp, N, cin, cout, stream);
  // message passing
  hipMemsetAsync(w.agg, 0, (size_t)N * cout * sizeof(float), stream);
  hipMemsetAsync(w.deg, 0, (size_t)N * sizeof(float), stream);
  const int* src = edges;
  const int* dst = edges + E;
  edge_w_kernel<<<(E + 255) / 256, 256, 0, stream>>>(ev, mu, sg, w.wbuf, E);
  long long te = (long long)E * cout;
  unsigned gb = (unsigned)((te + 255) / 256);
  switch (cout) {
    case 32:  edge_msg_t<32><<<gb, 256, 0, stream>>>(w.xg, src, dst, w.wbuf, w.agg, E); break;
    case 64:  edge_msg_t<64><<<gb, 256, 0, stream>>>(w.xg, src, dst, w.wbuf, w.agg, E); break;
    default:  edge_msg_t<128><<<gb, 256, 0, stream>>>(w.xg, src, dst, w.wbuf, w.agg, E); break;
  }
  deg_kernel<<<(E + 255) / 256, 256, 0, stream>>>(dst, w.deg, E);
  long long tn = (long long)N * cout;
  finalize_kernel<<<(unsigned)((tn + 255) / 256), 256, 0, stream>>>(w.agg, w.deg, w.rootp,
                                                                    bias, out, N, cout);
}

extern "C" void kernel_launch(void* const* d_in, const int* in_sizes, int n_in,
                              void* d_out, int out_size, void* d_ws, size_t ws_size,
                              hipStream_t stream) {
  (void)in_sizes; (void)n_in; (void)out_size; (void)ws_size;

  const float* data = (const float*)d_in[0];
  const int cins[10]  = {32, 32, 32, 64, 64, 128, 192, 64, 96, 32};
  const int couts[10] = {32, 32, 64, 64, 128, 128, 64, 64, 32, 32};
  const float *G[10], *MU[10], *SG[10], *RT[10], *BS[10];
  for (int i = 0; i < 10; ++i) {
    G[i]  = (const float*)d_in[1 + i * 5 + 0];
    MU[i] = (const float*)d_in[1 + i * 5 + 1];
    SG[i] = (const float*)d_in[1 + i * 5 + 2];
    RT[i] = (const float*)d_in[1 + i * 5 + 3];
    BS[i] = (const float*)d_in[1 + i * 5 + 4];
  }
  const float* fcw = (const float*)d_in[51];
  const float* fcb = (const float*)d_in[52];
  const int*   e7  = (const int*)d_in[53];
  const float* p7  = (const float*)d_in[54];
  const int*   e6  = (const int*)d_in[55];
  const float* p6  = (const float*)d_in[56];
  const int*   e5  = (const int*)d_in[57];
  const float* p5  = (const float*)d_in[58];
  const int* pool76 = (const int*)d_in[59];
  const int* pool65 = (const int*)d_in[60];
  const int* up6    = (const int*)d_in[61];
  const int* up7    = (const int*)d_in[62];

  // workspace carve-up (float units, 256B aligned regions)
  float* base = (float*)d_ws;
  size_t off = 0;
  auto carve = [&](size_t nf) { float* p = base + off; off += (nf + 63) & ~(size_t)63; return p; };
  const size_t NB = (size_t)NV7 * 96;          // max node-feature / xg element count
  float* bufA  = carve(NB);
  float* bufB  = carve(NB);
  GmmWs ws;
  ws.xg    = carve(NB);
  ws.rootp = carve((size_t)NV7 * 32);
  ws.agg   = carve((size_t)NV7 * 32);
  ws.deg   = carve((size_t)NV7);
  ws.wbuf  = carve((size_t)NE7 * 3);
  float* sk7 = carve((size_t)NV7 * 32);
  float* sk6 = carve((size_t)NV6 * 64);
  ws.xbf   = (__bf16*)carve(NB / 2 + 64);      // NB bf16 elements
  ws.gfrag = (__bf16*)carve(64 * 1024);        // >= 24 tiles * 6 ksteps * 32 lanes * 16 bf16
  ws.rfrag = (__bf16*)carve(64 * 1024);

  // ---- encoder block 0 (level 7, 32ch) ----
  run_gmm_layer(data, NV7, cins[0], couts[0], G[0], MU[0], SG[0], RT[0], BS[0], e7, p7, NE7, bufA, ws, stream);
  run_gmm_layer(bufA, NV7, cins[1], couts[1], G[1], MU[1], SG[1], RT[1], BS[1], e7, p7, NE7, bufB, ws, stream);
  hipMemcpyAsync(sk7, bufB, (size_t)NV7 * 32 * sizeof(float), hipMemcpyDeviceToDevice, stream);
  {
    long long t = (long long)NV6 * 32;
    hexpool_kernel<<<(unsigned)((t + 255) / 256), 256, 0, stream>>>(bufB, pool76, bufA, NV6, 32);
  }
  // ---- encoder block 1 (level 6, 64ch) ----
  run_gmm_layer(bufA, NV6, cins[2], couts[2], G[2], MU[2], SG[2], RT[2], BS[2], e6, p6, NE6, bufB, ws, stream);
  run_gmm_layer(bufB, NV6, cins[3], couts[3], G[3], MU[3], SG[3], RT[3], BS[3], e6, p6, NE6, bufA, ws, stream);
  hipMemcpyAsync(sk6, bufA, (size_t)NV6 * 64 * sizeof(float), hipMemcpyDeviceToDevice, stream);
  {
    long long t = (long long)NV5 * 64;
    hexpool_kernel<<<(unsigned)((t + 255) / 256), 256, 0, stream>>>(bufA, pool65, bufB, NV5, 64);
  }
  // ---- encoder block 2 (level 5, 128ch) ----
  run_gmm_layer(bufB, NV5, cins[4], couts[4], G[4], MU[4], SG[4], RT[4], BS[4], e5, p5, NE5, bufA, ws, stream);
  run_gmm_layer(bufA, NV5, cins[5], couts[5], G[5], MU[5], SG[5], RT[5], BS[5], e5, p5, NE5, bufB, ws, stream);
  // ---- decoder block 0 (unpool 5->6, concat skip6: 128+64=192 -> 64) ----
  {
    long long t = (long long)NV6 * 192;
    unpool_concat_kernel<<<(unsigned)((t + 255) / 256), 256, 0, stream>>>(bufB, up6, sk6, bufA, NV5, NV6, 128, 64);
  }
  run_gmm_layer(bufA, NV6, cins[6], couts[6], G[6], MU[6], SG[6], RT[6], BS[6], e6, p6, NE6, bufB, ws, stream);
  run_gmm_layer(bufB, NV6, cins[7], couts[7], G[7], MU[7], SG[7], RT[7], BS[7], e6, p6, NE6, bufA, ws, stream);
  // ---- decoder block 1 (unpool 6->7, concat skip7: 64+32=96 -> 32) ----
  {
    long long t = (long long)NV7 * 96;
    unpool_concat_kernel<<<(unsigned)((t + 255) / 256), 256, 0, stream>>>(bufA, up7, sk7, bufB, NV6, NV7, 64, 32);
  }
  run_gmm_layer(bufB, NV7, cins[8], couts[8], G[8], MU[8], SG[8], RT[8], BS[8], e7, p7, NE7, bufA, ws, stream);
  run_gmm_layer(bufA, NV7, cins[9], couts[9], G[9], MU[9], SG[9], RT[9], BS[9], e7, p7, NE7, bufB, ws, stream);
  // ---- final FC + log_softmax ----
  fc_lsm_kernel<<<(NV7 + 255) / 256, 256, 0, stream>>>(bufB, fcw, fcb, (float*)d_out, NV7);
}
